// MoELayer_43310450213489
// MI455X (gfx1250) — matrix operand order repaired and checked by reading, not compiled, not run on hardware
//
#include <hip/hip_runtime.h>
#include <hip/hip_bf16.h>

typedef __bf16 bf16_t;
typedef __attribute__((ext_vector_type(16))) __bf16 v16bf;
typedef __attribute__((ext_vector_type(8)))  float   v8f;

#define N_TOK   8192
#define D_MODEL 1024
#define N_EXP   8
#define H_DIM   2048
#define TILE_M  32
#define HSLAB   1024                     /* h computed in 2 column slabs */
#define MAX_TILES (N_TOK / TILE_M)       /* 256 */

// Branchless tanh-GELU: tanh(u) = 1 - 2/(exp(2u)+1) via v_exp_f32.
__device__ __forceinline__ float gelu_f(float v) {
    const float u = v * (0.7978845608f + 0.0356774081f * v * v);
    const float t = 1.0f - 2.0f / (__expf(2.0f * u) + 1.0f);
    return 0.5f * v * (1.0f + t);
}

// ---------------------------------------------------------------------------
// Router: 1 wave (32 lanes) per token. logits = x[n]·gate_w, softmax, top-2,
// renormalize; build per-expert token lists + aux-loss accumulators.
// ---------------------------------------------------------------------------
__global__ __launch_bounds__(256) void moe_router_kernel(
    const float* __restrict__ x, const float* __restrict__ gw,
    float* __restrict__ probs_out, float* __restrict__ tidx_out,
    float* __restrict__ tprob_out,
    int* __restrict__ expCount, float* __restrict__ probSum,
    int* __restrict__ tokId, float* __restrict__ tokW)
{
    const int wave = threadIdx.x >> 5;
    const int lane = threadIdx.x & 31;
    const int n    = blockIdx.x * 8 + wave;

    float acc[N_EXP];
#pragma unroll
    for (int e = 0; e < N_EXP; ++e) acc[e] = 0.0f;

    const float* xr = x + (size_t)n * D_MODEL;
    for (int d = lane; d < D_MODEL; d += 32) {
        const float xv = xr[d];
        const float* g = gw + (size_t)d * N_EXP;
#pragma unroll
        for (int e = 0; e < N_EXP; ++e) acc[e] += xv * g[e];
    }
#pragma unroll
    for (int e = 0; e < N_EXP; ++e) {
#pragma unroll
        for (int off = 16; off > 0; off >>= 1)
            acc[e] += __shfl_xor(acc[e], off, 32);
    }

    if (lane == 0) {
        float mx = acc[0];
#pragma unroll
        for (int e = 1; e < N_EXP; ++e) mx = fmaxf(mx, acc[e]);
        float p[N_EXP];
        float se = 0.0f;
#pragma unroll
        for (int e = 0; e < N_EXP; ++e) { p[e] = __expf(acc[e] - mx); se += p[e]; }
        const float inv = 1.0f / se;
#pragma unroll
        for (int e = 0; e < N_EXP; ++e) {
            p[e] *= inv;
            probs_out[(size_t)n * N_EXP + e] = p[e];
            atomicAdd(&probSum[e], p[e]);
        }
        int i0 = 0;
#pragma unroll
        for (int e = 1; e < N_EXP; ++e) if (p[e] > p[i0]) i0 = e;
        int i1 = (i0 == 0) ? 1 : 0;
#pragma unroll
        for (int e = 0; e < N_EXP; ++e) if (e != i0 && p[e] > p[i1]) i1 = e;
        const float s2 = p[i0] + p[i1];
        const float w0 = p[i0] / s2;
        const float w1 = p[i1] / s2;

        tidx_out[(size_t)n * 2 + 0]  = (float)i0;
        tidx_out[(size_t)n * 2 + 1]  = (float)i1;
        tprob_out[(size_t)n * 2 + 0] = w0;
        tprob_out[(size_t)n * 2 + 1] = w1;

        int s0 = atomicAdd(&expCount[i0], 1);
        tokId[i0 * N_TOK + s0] = n;
        tokW [i0 * N_TOK + s0] = w0;
        int s1 = atomicAdd(&expCount[i1], 1);
        tokId[i1 * N_TOK + s1] = n;
        tokW [i1 * N_TOK + s1] = w1;
    }
}

// ---------------------------------------------------------------------------
// Aux loss: E * sum_e (count_e/N) * (probSum_e/N)
// ---------------------------------------------------------------------------
__global__ void moe_aux_kernel(const int* __restrict__ expCount,
                               const float* __restrict__ probSum,
                               float* __restrict__ aux)
{
    if (threadIdx.x == 0 && blockIdx.x == 0) {
        float a = 0.0f;
#pragma unroll
        for (int e = 0; e < N_EXP; ++e)
            a += ((float)expCount[e] / (float)N_TOK) * (probSum[e] / (float)N_TOK);
        aux[0] = (float)N_EXP * a;
    }
}

// ---------------------------------------------------------------------------
// One-time repack of f32 weights into bf16 WMMA-fragment order:
//   pw[((e*NT + nt)*KT + kt)*32 + lane][j]  holds W[e][kt*32 + (lane>>4)*16 + j]
//                                                    [nt*16 + (lane&15)]
// so a B fragment is one contiguous, 32B-aligned v16bf per lane.
// ---------------------------------------------------------------------------
__global__ __launch_bounds__(256) void pack_weights_kernel(
    const float* __restrict__ w, bf16_t* __restrict__ pw, int K, int Nw)
{
    const int NT = Nw >> 4, KT = K >> 5;
    const size_t total = (size_t)N_EXP * NT * KT * 32;
    size_t t = (size_t)blockIdx.x * 256 + threadIdx.x;
    if (t >= total) return;
    const int lane = (int)(t & 31);
    size_t q = t >> 5;
    const int kt = (int)(q % KT); q /= KT;
    const int nt = (int)(q % NT);
    const int e  = (int)(q / NT);
    const int nl = lane & 15, kh = lane >> 4;
    const float* src = w + (size_t)e * K * Nw
                         + (size_t)(kt * 32 + kh * 16) * Nw + nt * 16 + nl;
    bf16_t* dst = pw + t * 16;
#pragma unroll
    for (int j = 0; j < 16; ++j) dst[j] = (bf16_t)src[(size_t)j * Nw];
}

// ---------------------------------------------------------------------------
// Fused expert FFN. One block = one expert e, one 32-token tile.
//   h = gelu(X @ w1[e]) computed in two 1024-col slabs (LDS reuse);
//   y accumulators (16 x v8f = 128 VGPR) persist across slabs;
//   out[tok] += gate * y via global_atomic_add_f32.
// 8 waves; each B fragment is reused for both 16-row M-subtiles (2 WMMA/load).
// ---------------------------------------------------------------------------
template <bool PACKED>
__global__ __launch_bounds__(256) void moe_ffn_kernel(
    const float* __restrict__ x,
    const float* __restrict__ w1, const float* __restrict__ w2,
    const bf16_t* __restrict__ pw1, const bf16_t* __restrict__ pw2,
    float* __restrict__ out,
    const int* __restrict__ expCount, const int* __restrict__ tokId,
    const float* __restrict__ tokW)
{
    __shared__ bf16_t lds_x[TILE_M][D_MODEL + 8];   // ~64.5 KB
    __shared__ bf16_t lds_h[TILE_M][HSLAB + 8];     // ~64.5 KB
    __shared__ int    sTok[TILE_M];
    __shared__ float  sW[TILE_M];

    const int e    = blockIdx.x / MAX_TILES;
    const int tile = blockIdx.x % MAX_TILES;
    const int cnt  = expCount[e];
    if (tile * TILE_M >= cnt) return;               // uniform block exit

    if (threadIdx.x < TILE_M) {
        const int r = tile * TILE_M + threadIdx.x;
        if (r < cnt) {
            sTok[threadIdx.x] = tokId[e * N_TOK + r];
            sW [threadIdx.x]  = tokW [e * N_TOK + r];
        } else {                                    // pad: valid addr, zero gate
            sTok[threadIdx.x] = 0;
            sW [threadIdx.x]  = 0.0f;
        }
    }
    __syncthreads();

    for (int i = threadIdx.x; i < TILE_M * D_MODEL; i += 256) {
        const int r = i >> 10;
        const int c = i & (D_MODEL - 1);
        lds_x[r][c] = (bf16_t)x[(size_t)sTok[r] * D_MODEL + c];
    }
    __syncthreads();

    const int wave = threadIdx.x >> 5;
    const int lane = threadIdx.x & 31;
    const int kh   = lane >> 4;
    const int nl   = lane & 15;

    const float*  W1  = w1  + (size_t)e * D_MODEL * H_DIM;
    const float*  W2  = w2  + (size_t)e * H_DIM * D_MODEL;
    const bf16_t* PW1 = pw1 + (size_t)e * D_MODEL * H_DIM;
    const bf16_t* PW2 = pw2 + (size_t)e * H_DIM * D_MODEL;

    v8f acc[16] = {};                               // phase-2 accumulators
    const int dcol = wave * 128;                    // this wave's D columns

    for (int pass = 0; pass < 2; ++pass) {
        // ---- Phase 1: h slab = gelu(X @ w1[:, pass*1024 .. +1024)) ----
        for (int nt = 0; nt < 8; ++nt) {
            const int hcol = pass * HSLAB + wave * 128 + nt * 16;  // global col
            const int lcol = wave * 128 + nt * 16;                 // slab col
            v8f c0 = {}, c1 = {};
            for (int kb = 0; kb < D_MODEL; kb += 32) {
                v16bf b;
                if (PACKED) {
                    const bf16_t* fp = PW1 +
                        ((((size_t)(hcol >> 4)) * (D_MODEL / 32) + (kb >> 5)) * 32
                         + lane) * 16;
                    b = *(const v16bf*)fp;          // two global_load_b128
                } else {
                    const float* bp = W1 + (size_t)(kb + kh * 16) * H_DIM + hcol + nl;
#pragma unroll
                    for (int j = 0; j < 16; ++j) b[j] = (bf16_t)bp[(size_t)j * H_DIM];
                }
                v16bf a0, a1;
                const bf16_t* ap0 = &lds_x[nl]     [kb + kh * 16];
                const bf16_t* ap1 = &lds_x[16 + nl][kb + kh * 16];
#pragma unroll
                for (int j = 0; j < 8; ++j) {
                    a0[j] = ap0[j]; a0[8 + j] = ap0[8 + j];
                    a1[j] = ap1[j]; a1[8 + j] = ap1[8 + j];
                }
                c0 = __builtin_amdgcn_wmma_f32_16x16x32_bf16(
                         false, a0, false, b, (short)0, c0, false, false);
                c1 = __builtin_amdgcn_wmma_f32_16x16x32_bf16(
                         false, a1, false, b, (short)0, c1, false, false);
            }
#pragma unroll
            for (int r = 0; r < 8; ++r) {
                lds_h[r + kh * 8]     [lcol + nl] = (bf16_t)gelu_f(c0[r]);
                lds_h[16 + r + kh * 8][lcol + nl] = (bf16_t)gelu_f(c1[r]);
            }
        }
        __syncthreads();

        // ---- Phase 2 partial: y += h_slab @ w2[pass*1024 .. +1024, :] ----
        for (int nt = 0; nt < 8; ++nt) {
            v8f c0 = acc[nt * 2 + 0];
            v8f c1 = acc[nt * 2 + 1];
            for (int kb = 0; kb < HSLAB; kb += 32) {
                const int kg = pass * HSLAB + kb;   // global K row in w2
                v16bf b;
                if (PACKED) {
                    const bf16_t* fp = PW2 +
                        ((((size_t)((dcol + nt * 16) >> 4)) * (H_DIM / 32) + (kg >> 5)) * 32
                         + lane) * 16;
                    b = *(const v16bf*)fp;
                } else {
                    const float* bp = W2 + (size_t)(kg + kh * 16) * D_MODEL
                                         + dcol + nt * 16 + nl;
#pragma unroll
                    for (int j = 0; j < 16; ++j) b[j] = (bf16_t)bp[(size_t)j * D_MODEL];
                }
                v16bf a0, a1;
                const bf16_t* ap0 = &lds_h[nl]     [kb + kh * 16];
                const bf16_t* ap1 = &lds_h[16 + nl][kb + kh * 16];
#pragma unroll
                for (int j = 0; j < 8; ++j) {
                    a0[j] = ap0[j]; a0[8 + j] = ap0[8 + j];
                    a1[j] = ap1[j]; a1[8 + j] = ap1[8 + j];
                }
                c0 = __builtin_amdgcn_wmma_f32_16x16x32_bf16(
                         false, a0, false, b, (short)0, c0, false, false);
                c1 = __builtin_amdgcn_wmma_f32_16x16x32_bf16(
                         false, a1, false, b, (short)0, c1, false, false);
            }
            acc[nt * 2 + 0] = c0;
            acc[nt * 2 + 1] = c1;
        }
        __syncthreads();                            // lds_h reused next pass
    }

    // ---- Scatter: out[tok] += gate * y ----
#pragma unroll
    for (int nt = 0; nt < 8; ++nt) {
#pragma unroll
        for (int mi = 0; mi < 2; ++mi) {
            const v8f c = acc[nt * 2 + mi];
#pragma unroll
            for (int r = 0; r < 8; ++r) {
                const int m = mi * 16 + kh * 8 + r;
                atomicAdd(&out[(size_t)sTok[m] * D_MODEL + dcol + nt * 16 + nl],
                          sW[m] * c[r]);
            }
        }
    }
}

// ---------------------------------------------------------------------------
extern "C" void kernel_launch(void* const* d_in, const int* in_sizes, int n_in,
                              void* d_out, int out_size, void* d_ws, size_t ws_size,
                              hipStream_t stream)
{
    const float* x  = (const float*)d_in[0];
    const float* gw = (const float*)d_in[1];
    const float* w1 = (const float*)d_in[2];
    const float* w2 = (const float*)d_in[3];

    float* out   = (float*)d_out;                       // [N, D]
    float* auxp  = out   + (size_t)N_TOK * D_MODEL;     // [1]
    float* probs = auxp  + 1;                           // [N, E]
    float* tidx  = probs + (size_t)N_TOK * N_EXP;       // [N, 2]
    float* tprb  = tidx  + (size_t)N_TOK * 2;           // [N, 2]

    int*   expCount = (int*)d_ws;                       // [8]
    float* probSum  = (float*)d_ws + 8;                 // [8]
    int*   tokId    = (int*)d_ws + 16;                  // [8][8192]
    float* tokW     = (float*)d_ws + 16 + N_EXP * N_TOK;

    const size_t headerBytes = (size_t)(16 + 2 * N_EXP * N_TOK) * 4;  // 524,352 (32B-mult)
    const size_t pwElems     = (size_t)N_EXP * D_MODEL * H_DIM;       // per weight tensor
    const size_t needBytes   = headerBytes + 2 * pwElems * sizeof(bf16_t);
    const bool   packed      = (ws_size >= needBytes);
    bf16_t* pw1 = (bf16_t*)((char*)d_ws + headerBytes);
    bf16_t* pw2 = pw1 + pwElems;

    hipMemsetAsync(d_out, 0, (size_t)N_TOK * D_MODEL * sizeof(float), stream);
    hipMemsetAsync(d_ws, 0, 64 * sizeof(float), stream);

    moe_router_kernel<<<N_TOK / 8, 256, 0, stream>>>(
        x, gw, probs, tidx, tprb, expCount, probSum, tokId, tokW);
    moe_aux_kernel<<<1, 32, 0, stream>>>(expCount, probSum, auxp);

    if (packed) {
        const int pk1Blocks = (int)(((size_t)N_EXP * (H_DIM / 16) * (D_MODEL / 32) * 32 + 255) / 256);
        const int pk2Blocks = (int)(((size_t)N_EXP * (D_MODEL / 16) * (H_DIM / 32) * 32 + 255) / 256);
        pack_weights_kernel<<<pk1Blocks, 256, 0, stream>>>(w1, pw1, D_MODEL, H_DIM);
        pack_weights_kernel<<<pk2Blocks, 256, 0, stream>>>(w2, pw2, H_DIM, D_MODEL);
        moe_ffn_kernel<true><<<N_EXP * MAX_TILES, 256, 0, stream>>>(
            x, w1, w2, pw1, pw2, out, expCount, tokId, tokW);
    } else {
        moe_ffn_kernel<false><<<N_EXP * MAX_TILES, 256, 0, stream>>>(
            x, w1, w2, pw1, pw2, out, expCount, tokId, tokW);
    }
}
